// PNA_Custom_25168508355336
// MI455X (gfx1250) — matrix operand order repaired and checked by reading, not compiled, not run on hardware
//
#include <hip/hip_runtime.h>
#include <stdint.h>

// ---------------------------------------------------------------------------
// CDNA5 (gfx1250) wave32 WMMA helpers: v_wmma_f32_16x16x32_bf16
// ---------------------------------------------------------------------------
typedef __attribute__((ext_vector_type(16))) __bf16 v16bf;
typedef __attribute__((ext_vector_type(8)))  float  v8f;

union Frag16 { v16bf v; uint32_t u[8]; };

__device__ __forceinline__ uint32_t f2bf1(float f) {
  union { float f; uint32_t u; } x{f};
  uint32_t r = x.u + 0x7FFFu + ((x.u >> 16) & 1u);   // RNE truncate to bf16
  return r >> 16;
}
__device__ __forceinline__ uint32_t pack2bf(float lo, float hi) {
#if defined(__gfx1250__) && __has_builtin(__builtin_amdgcn_cvt_pk_bf16_f32)
  auto r = __builtin_amdgcn_cvt_pk_bf16_f32(lo, hi);   // v_cvt_pk_bf16_f32
  return __builtin_bit_cast(uint32_t, r);
#else
  return f2bf1(lo) | (f2bf1(hi) << 16);
#endif
}
// order-preserving float<->uint for atomicMin/Max on u32
__device__ __forceinline__ uint32_t fenc(float f) {
  union { float f; uint32_t u; } x{f};
  return (x.u & 0x80000000u) ? ~x.u : (x.u | 0x80000000u);
}
__device__ __forceinline__ float fdec(uint32_t v) {
  union { float f; uint32_t u; } x;
  x.u = (v & 0x80000000u) ? (v ^ 0x80000000u) : ~v;
  return x.f;
}

// A fragment (16x32 bf16, MxK) from LDS stored row-major as bf16 pairs.
// ISA 7.12.2: lane m=l&15, half=l>>4; reg v -> K pair base (v<4 ? 2v : 16+2(v-4)) + 8*half
__device__ __forceinline__ Frag16 load_a(const uint32_t* lds, int rowBase,
                                         int strideP, int kp0) {
  Frag16 f;
  const int lane = threadIdx.x & 31;
  const int m = lane & 15, lh = lane >> 4;
#pragma unroll
  for (int r = 0; r < 8; ++r) {
    const int p = (r < 4 ? r : 8 + (r - 4)) + 4 * lh;   // pair units
    f.u[r] = lds[(rowBase + m) * strideP + kp0 + p];
  }
  return f;
}
// B fragment (32x16 bf16, KxN) from global weights stored TRANSPOSED:
// wT row n holds K bf16 (as K/2 uint pairs). lane n=l&15, half=l>>4; reg v -> K pair base (v<4 ? 2v : 16+2(v-4)) + 8*half
__device__ __forceinline__ Frag16 load_b(const uint32_t* __restrict__ wT, int nBase,
                                         int strideP, int kp0) {
  Frag16 f;
  const int lane = threadIdx.x & 31;
  const int n = lane & 15, lh = lane >> 4;
#pragma unroll
  for (int r = 0; r < 8; ++r) {
    f.u[r] = wT[(size_t)(nBase + n) * strideP + kp0 + (r < 4 ? r : 8 + (r - 4)) + 4 * lh];
  }
  return f;
}
__device__ __forceinline__ v8f wmma_bf16(Frag16 a, Frag16 b, v8f c) {
  return __builtin_amdgcn_wmma_f32_16x16x32_bf16(false, a.v, false, b.v,
                                                 (short)0, c, false, false);
}

// ---------------------------------------------------------------------------
// Small utility kernels
// ---------------------------------------------------------------------------
__global__ void k_fill_u(uint32_t* p, uint32_t v, size_t n) {
  size_t i  = (size_t)blockIdx.x * blockDim.x + threadIdx.x;
  size_t st = (size_t)gridDim.x * blockDim.x;
  for (; i < n; i += st) p[i] = v;
}

__global__ void k_count(const int* __restrict__ dst, float* counts, int E) {
  int i = blockIdx.x * blockDim.x + threadIdx.x;
  if (i < E) atomicAdd(&counts[dst[i]], 1.0f);
}

__global__ void k_sumlog(const float* __restrict__ counts, float* acc, int N) {
  __shared__ float sm[256];
  float s = 0.f;
  for (int i = blockIdx.x * blockDim.x + threadIdx.x; i < N;
       i += gridDim.x * blockDim.x)
    s += log1pf(counts[i]);
  sm[threadIdx.x] = s;
  __syncthreads();
  for (int o = 128; o > 0; o >>= 1) {
    if ((int)threadIdx.x < o) sm[threadIdx.x] += sm[threadIdx.x + o];
    __syncthreads();
  }
  if (threadIdx.x == 0) atomicAdd(acc, sm[0]);
}

__global__ void k_scalers(const float* __restrict__ counts,
                          const float* __restrict__ sumlog, float* degv,
                          float* ampv, float* attv, float* hasv, int N) {
  int i = blockIdx.x * blockDim.x + threadIdx.x;
  if (i >= N) return;
  const float avg = sumlog[0] / (float)N;
  const float c  = counts[i];
  const float dg = fmaxf(c, 1.0f);
  const float ld = logf(dg + 1.0f);
  degv[i] = dg;
  ampv[i] = ld / avg;
  attv[i] = avg / ld;
  hasv[i] = (c > 0.f) ? 1.f : 0.f;
}

// Transpose-convert fp32 weight block [K rows x Nc cols] (row stride `stride`,
// starting row rowOff) into bf16 wT: row n holds K bf16 packed as uint pairs.
__global__ void k_convT(const float* __restrict__ src, uint32_t* __restrict__ dstT,
                        int K, int Nc, int rowOff, int stride) {
  int i = blockIdx.x * blockDim.x + threadIdx.x;
  const int kp = K / 2;
  if (i >= Nc * kp) return;
  const int n = i / kp, p = i % kp;
  const float lo = src[(size_t)(rowOff + 2 * p) * stride + n];
  const float hi = src[(size_t)(rowOff + 2 * p + 1) * stride + n];
  dstT[(size_t)n * kp + p] = pack2bf(lo, hi);
}

// ---------------------------------------------------------------------------
// Edge kernel: h = [x_dst | x_src] @ pre_w + pre_b, fused segment aggregation.
// EDGES = 64 (FIN=128) or 128 (FIN=64) edges/block; one 16-edge tile per wave.
// ---------------------------------------------------------------------------
template <int FIN>
__global__ __launch_bounds__(FIN == 128 ? 128 : 256) void k_edge(
    const float* __restrict__ hin, const int* __restrict__ src,
    const int* __restrict__ dst, const uint32_t* __restrict__ preT,
    const float* __restrict__ pb, float* __restrict__ s1,
    float* __restrict__ s2, uint32_t* __restrict__ mnU,
    uint32_t* __restrict__ mxU, int E) {
  constexpr int KP = FIN;                       // A row = 2*FIN bf16 = FIN pairs
  constexpr int EDGES = (FIN == 128) ? 64 : 128;
  __shared__ uint32_t aLds[EDGES * KP];
  __shared__ int dstLds[EDGES];

  const int t = threadIdx.x;
  const int blockE0 = blockIdx.x * EDGES;
  const bool full = (blockE0 + EDGES <= E);     // block-uniform fast path
  {  // stage gathered [x_dst | x_src] rows as bf16 pairs; 2 threads per edge
    const int eLocal = t >> 1;
    const int part   = t & 1;  // 0: dst half, 1: src half
    const int e = blockE0 + eLocal;
    uint32_t* out = aLds + eLocal * KP + part * (FIN / 2);
    if (full || e < E) {
      const int node = part ? src[e] : dst[e];
      if (part == 0) dstLds[eLocal] = dst[e];
      const float2* row = (const float2*)(hin + (size_t)node * FIN);
#pragma unroll
      for (int i = 0; i < FIN / 2; ++i) {
        float2 v = row[i];
        out[i] = pack2bf(v.x, v.y);
      }
    } else {
      if (part == 0) dstLds[eLocal] = 0;
#pragma unroll
      for (int i = 0; i < FIN / 2; ++i) out[i] = 0u;
    }
  }
  __syncthreads();

  const int wave = t >> 5;
  const int lane = t & 31;
  const int lh   = lane >> 4;
  const int nIn  = lane & 15;
  const int rowBase = wave * 16;

#pragma unroll 1
  for (int ntile = 0; ntile < FIN / 16; ++ntile) {
    __builtin_prefetch(preT + (size_t)(ntile * 16) * KP, 0, 3);
    v8f acc = {};
#pragma unroll
    for (int kp0 = 0; kp0 < KP; kp0 += 16) {
      Frag16 a = load_a(aLds, rowBase, KP, kp0);
      Frag16 b = load_b(preT, ntile * 16, KP, kp0);
      acc = wmma_bf16(a, b, acc);
    }
    const int n = ntile * 16 + nIn;
    const float bias = pb[n];
    if (full) {
#pragma unroll
      for (int r = 0; r < 8; ++r) {
        const int m = r + 8 * lh;           // C layout: reg r -> row m (+8 hi half)
        const float h = acc[r] + bias;
        const size_t o = (size_t)dstLds[rowBase + m] * FIN + n;
        atomicAdd(&s1[o], h);
        atomicAdd(&s2[o], h * h);
        atomicMin(&mnU[o], fenc(h));
        atomicMax(&mxU[o], fenc(h));
      }
    } else {
#pragma unroll
      for (int r = 0; r < 8; ++r) {
        const int m = r + 8 * lh;
        if (blockE0 + rowBase + m < E) {
          const float h = acc[r] + bias;
          const size_t o = (size_t)dstLds[rowBase + m] * FIN + n;
          atomicAdd(&s1[o], h);
          atomicAdd(&s2[o], h * h);
          atomicMin(&mnU[o], fenc(h));
          atomicMax(&mxU[o], fenc(h));
        }
      }
    }
  }
}

// ---------------------------------------------------------------------------
// Node kernel: fused post_nn (factored amp/att), lin, skip, ELU.
// Block = 128 threads (4 waves) per 16-node tile; wave w -> out cols [16w,16w+16).
// u = hin@W0 + agg@W1 + amp*(agg@W2) + att*(agg@W3) + ob ; out = elu(u@lw + lb + skip)
// ---------------------------------------------------------------------------
template <int FIN, bool SKIP_GEMM>
__global__ __launch_bounds__(128) void k_node(
    const float* __restrict__ hin, const float* __restrict__ s1,
    const float* __restrict__ s2, const uint32_t* __restrict__ mnU,
    const uint32_t* __restrict__ mxU, const float* __restrict__ degv,
    const float* __restrict__ ampv, const float* __restrict__ attv,
    const float* __restrict__ hasv, const uint32_t* __restrict__ w0T,
    const uint32_t* __restrict__ waggT, const float* __restrict__ ob,
    const uint32_t* __restrict__ linT, const float* __restrict__ lb,
    const uint32_t* __restrict__ swT, const float* __restrict__ sb,
    float* __restrict__ out, int N) {
  constexpr int KPA = 2 * FIN;   // agg row: 4*FIN bf16 = 2*FIN pairs
  constexpr int KPH = FIN / 2;   // hin row: FIN bf16
  __shared__ uint32_t aggA[16 * KPA];
  __shared__ uint32_t hinA[16 * KPH];
  __shared__ uint32_t uA[16 * 32];      // u tile 16x64 bf16
  __shared__ float ampL[16], attL[16];

  const int node0 = blockIdx.x * 16;
  const bool fullB = (node0 + 16 <= N);
  const int t = threadIdx.x;
  if (t < 16) {
    const int nd = node0 + t;
    ampL[t] = (fullB || nd < N) ? ampv[nd] : 0.f;
    attL[t] = (fullB || nd < N) ? attv[nd] : 0.f;
  }
  // Build agg = [mean|min|max|std] and hin tiles (bf16) for 16 nodes.
  for (int idx = t; idx < 16 * (FIN / 2); idx += 128) {
    const int i  = idx / (FIN / 2);
    const int p  = idx % (FIN / 2);
    const int nd = node0 + i;
    uint32_t* row = aggA + i * KPA;
    if (fullB || nd < N) {
      const size_t o = (size_t)nd * FIN + 2 * p;
      const float dg = degv[nd];
      const float hs = hasv[nd];
      const float me0 = s1[o] / dg, me1 = s1[o + 1] / dg;
      const float va0 = s2[o] / dg - me0 * me0;
      const float va1 = s2[o + 1] / dg - me1 * me1;
      const float sd0 = sqrtf(fmaxf(va0, 0.f) + 1e-5f);
      const float sd1 = sqrtf(fmaxf(va1, 0.f) + 1e-5f);
      const float mn0 = (hs != 0.f) ? fdec(mnU[o]) : 0.f;
      const float mn1 = (hs != 0.f) ? fdec(mnU[o + 1]) : 0.f;
      const float mx0 = (hs != 0.f) ? fdec(mxU[o]) : 0.f;
      const float mx1 = (hs != 0.f) ? fdec(mxU[o + 1]) : 0.f;
      row[p]                   = pack2bf(me0, me1);
      row[(FIN / 2) + p]       = pack2bf(mn0, mn1);
      row[2 * (FIN / 2) + p]   = pack2bf(mx0, mx1);
      row[3 * (FIN / 2) + p]   = pack2bf(sd0, sd1);
      const float2 hv = *(const float2*)(hin + (size_t)nd * FIN + 2 * p);
      hinA[i * KPH + p] = pack2bf(hv.x, hv.y);
    } else {
      row[p] = row[(FIN / 2) + p] = row[2 * (FIN / 2) + p] =
          row[3 * (FIN / 2) + p] = 0u;
      hinA[i * KPH + p] = 0u;
    }
  }
  __syncthreads();

  const int wave = t >> 5;
  const int lane = t & 31;
  const int lh   = lane >> 4;
  const int nOut = wave * 16 + (lane & 15);

  v8f acc0 = {};
#pragma unroll
  for (int kp0 = 0; kp0 < KPH; kp0 += 16) {
    Frag16 a = load_a(hinA, 0, KPH, kp0);
    Frag16 b = load_b(w0T, wave * 16, KPH, kp0);
    acc0 = wmma_bf16(a, b, acc0);
  }
  v8f acc1 = {}, acc2 = {}, acc3 = {};
#pragma unroll 1
  for (int kp0 = 0; kp0 < KPA; kp0 += 16) {
    Frag16 a  = load_a(aggA, 0, KPA, kp0);
    Frag16 b1 = load_b(waggT, wave * 16, KPA, kp0);
    acc1 = wmma_bf16(a, b1, acc1);
    Frag16 b2 = load_b(waggT, 64 + wave * 16, KPA, kp0);
    acc2 = wmma_bf16(a, b2, acc2);
    Frag16 b3 = load_b(waggT, 128 + wave * 16, KPA, kp0);
    acc3 = wmma_bf16(a, b3, acc3);
  }
  const float biasO = ob[nOut];
  unsigned short* uH = (unsigned short*)uA;
#pragma unroll
  for (int r = 0; r < 8; ++r) {
    const int m = r + 8 * lh;
    const float u =
        acc0[r] + acc1[r] + ampL[m] * acc2[r] + attL[m] * acc3[r] + biasO;
    uH[m * 64 + nOut] = (unsigned short)f2bf1(u);
  }
  __syncthreads();

  v8f vac = {};
#pragma unroll
  for (int kp0 = 0; kp0 < 32; kp0 += 16) {
    Frag16 a = load_a(uA, 0, 32, kp0);
    Frag16 b = load_b(linT, wave * 16, 32, kp0);
    vac = wmma_bf16(a, b, vac);
  }
  if constexpr (SKIP_GEMM) {
    v8f sac = {};
#pragma unroll
    for (int kp0 = 0; kp0 < KPH; kp0 += 16) {
      Frag16 a = load_a(hinA, 0, KPH, kp0);
      Frag16 b = load_b(swT, wave * 16, KPH, kp0);
      sac = wmma_bf16(a, b, sac);
    }
    const float biasS = sb[nOut];
#pragma unroll
    for (int r = 0; r < 8; ++r) vac[r] += sac[r] + biasS;
  }
  const float biasL = lb[nOut];
#pragma unroll
  for (int r = 0; r < 8; ++r) {
    const int m  = r + 8 * lh;
    const int nd = node0 + m;
    if (fullB || nd < N) {
      float v = vac[r] + biasL;
      if constexpr (!SKIP_GEMM) v += hin[(size_t)nd * FIN + nOut];  // identity skip
      v = (v > 0.f) ? v : (expf(v) - 1.f);
      out[(size_t)nd * 64 + nOut] = v;
    }
  }
}

// ---------------------------------------------------------------------------
// Host orchestration
// ---------------------------------------------------------------------------
extern "C" void kernel_launch(void* const* d_in, const int* in_sizes, int n_in,
                              void* d_out, int out_size, void* d_ws,
                              size_t ws_size, hipStream_t stream) {
  (void)n_in; (void)out_size; (void)ws_size;
  const int N = in_sizes[0] / 128;
  const int E = in_sizes[1] / 2;

  const float* x   = (const float*)d_in[0];
  const int*   ei  = (const int*)d_in[1];
  const int*   src = ei;       // edge_index[0]
  const int*   dst = ei + E;   // edge_index[1]
  const float* pre_w[3]  = {(const float*)d_in[2],  (const float*)d_in[8],  (const float*)d_in[14]};
  const float* pre_b[3]  = {(const float*)d_in[3],  (const float*)d_in[9],  (const float*)d_in[15]};
  const float* post_w[3] = {(const float*)d_in[4],  (const float*)d_in[10], (const float*)d_in[16]};
  const float* post_b[3] = {(const float*)d_in[5],  (const float*)d_in[11], (const float*)d_in[17]};
  const float* lin_w[3]  = {(const float*)d_in[6],  (const float*)d_in[12], (const float*)d_in[18]};
  const float* lin_b[3]  = {(const float*)d_in[7],  (const float*)d_in[13], (const float*)d_in[19]};
  const float* skip_w = (const float*)d_in[20];
  const float* skip_b = (const float*)d_in[21];

  // workspace arena
  size_t off = 0;
  char* base = (char*)d_ws;
  auto alloc = [&](size_t bytes) -> void* {
    void* p = base + off;
    off += (bytes + 255) & ~(size_t)255;
    return p;
  };
  float*    counts = (float*)alloc((size_t)N * 4);
  float*    sumlog = (float*)alloc(4);
  float*    degv   = (float*)alloc((size_t)N * 4);
  float*    ampv   = (float*)alloc((size_t)N * 4);
  float*    attv   = (float*)alloc((size_t)N * 4);
  float*    hasv   = (float*)alloc((size_t)N * 4);
  float*    s1  = (float*)alloc((size_t)N * 128 * 4);
  float*    s2  = (float*)alloc((size_t)N * 128 * 4);
  uint32_t* mnU = (uint32_t*)alloc((size_t)N * 128 * 4);
  uint32_t* mxU = (uint32_t*)alloc((size_t)N * 128 * 4);
  float*    hA  = (float*)alloc((size_t)N * 64 * 4);
  float*    hB  = (float*)alloc((size_t)N * 64 * 4);
  const int FINs[3] = {128, 64, 64};
  uint32_t* preT[3]; uint32_t* w0T[3]; uint32_t* waggT[3]; uint32_t* linT[3];
  for (int l = 0; l < 3; ++l) {
    const int F = FINs[l];
    preT[l]  = (uint32_t*)alloc((size_t)F * F * 4);        // [F rows][F pairs]
    w0T[l]   = (uint32_t*)alloc((size_t)64 * (F / 2) * 4); // [64][F/2 pairs]
    waggT[l] = (uint32_t*)alloc((size_t)192 * 2 * F * 4);  // [192][2F pairs]
    linT[l]  = (uint32_t*)alloc((size_t)64 * 32 * 4);
  }
  uint32_t* sw0T = (uint32_t*)alloc((size_t)64 * 64 * 4);  // [64][64 pairs]

  // ---- degree stats ----
  k_fill_u<<<256, 256, 0, stream>>>((uint32_t*)counts, 0u, (size_t)N);
  k_fill_u<<<1, 64, 0, stream>>>((uint32_t*)sumlog, 0u, 1);
  k_count<<<(E + 255) / 256, 256, 0, stream>>>(dst, counts, E);
  k_sumlog<<<256, 256, 0, stream>>>(counts, sumlog, N);
  k_scalers<<<(N + 255) / 256, 256, 0, stream>>>(counts, sumlog, degv, ampv,
                                                 attv, hasv, N);
  // ---- weight conversion (fp32 -> transposed bf16) ----
  for (int l = 0; l < 3; ++l) {
    const int F = FINs[l];
    int n;
    n = F * F;  // preT: K=2F, Nc=F
    k_convT<<<(n + 255) / 256, 256, 0, stream>>>(pre_w[l], preT[l], 2 * F, F, 0, F);
    n = 64 * (F / 2);  // w0T: K=F, Nc=64 (rows 0..F of post_w)
    k_convT<<<(n + 255) / 256, 256, 0, stream>>>(post_w[l], w0T[l], F, 64, 0, 64);
    n = 64 * 2 * F;    // waggT blocks: K=4F, Nc=64, rows F+4F*b
    for (int b = 0; b < 3; ++b)
      k_convT<<<(n + 255) / 256, 256, 0, stream>>>(
          post_w[l], waggT[l] + (size_t)64 * b * 2 * F, 4 * F, 64, F + 4 * F * b, 64);
    n = 64 * 32;       // linT: K=64, Nc=64
    k_convT<<<(n + 255) / 256, 256, 0, stream>>>(lin_w[l], linT[l], 64, 64, 0, 64);
  }
  {
    int n = 64 * 64;   // sw0T: K=128, Nc=64
    k_convT<<<(n + 255) / 256, 256, 0, stream>>>(skip_w, sw0T, 128, 64, 0, 64);
  }

  const int nodeBlocks = (N + 15) / 16;
  const float* hin[3] = {x, hA, hB};
  float*       hout[3] = {hA, hB, (float*)d_out};

  for (int l = 0; l < 3; ++l) {
    const int F = FINs[l];
    const size_t agg_n = (size_t)N * F;
    k_fill_u<<<1024, 256, 0, stream>>>((uint32_t*)s1, 0u, agg_n);
    k_fill_u<<<1024, 256, 0, stream>>>((uint32_t*)s2, 0u, agg_n);
    k_fill_u<<<1024, 256, 0, stream>>>(mnU, 0xFFFFFFFFu, agg_n);
    k_fill_u<<<1024, 256, 0, stream>>>(mxU, 0u, agg_n);
    if (F == 128) {
      k_edge<128><<<(E + 63) / 64, 128, 0, stream>>>(
          hin[l], src, dst, preT[l], pre_b[l], s1, s2, mnU, mxU, E);
      k_node<128, true><<<nodeBlocks, 128, 0, stream>>>(
          hin[l], s1, s2, mnU, mxU, degv, ampv, attv, hasv, w0T[l], waggT[l],
          post_b[l], linT[l], lin_b[l], sw0T, skip_b, hout[l], N);
    } else {
      k_edge<64><<<(E + 127) / 128, 256, 0, stream>>>(
          hin[l], src, dst, preT[l], pre_b[l], s1, s2, mnU, mxU, E);
      k_node<64, false><<<nodeBlocks, 128, 0, stream>>>(
          hin[l], s1, s2, mnU, mxU, degv, ampv, attv, hasv, w0T[l], waggT[l],
          post_b[l], linT[l], lin_b[l], nullptr, nullptr, hout[l], N);
    }
  }
}
